// DeepGCN_66494683677236
// MI455X (gfx1250) — compile-verified, matrix operand order checked by reading, not run on hardware
//
#include <hip/hip_runtime.h>

// ---------------------------------------------------------------------------
// DeepGCN forward for MI455X (gfx1250, wave32, WMMA).
// GEMMs: bf16 WMMA (f32 accumulate); A tile staged in LDS per block via
// global_load_async_to_lds_b128 (ASYNCcnt), B pre-packed into WMMA fragments
// (L2-resident), 5 M-tiles per wave; A fragments preloaded per k-step so the
// 5 WMMAs issue back-to-back without per-WMMA dscnt stalls.
// SPMM: L2-resident gather + global_atomic_add_f32 scatter.
// PairNorm: two coalesced reductions + fused norm/relu/residual/bf16 repack.
// ---------------------------------------------------------------------------

#define NNODES 50000
#define DFEAT  512
#define DCLS   256
#define NEDGES 1600000

#define MTILES 5
#define MROWS  80          // 50000 = 625 * 80, exact tiling

typedef __attribute__((ext_vector_type(16))) __bf16 v16bf;
typedef __attribute__((ext_vector_type(8)))  __bf16 v8bf;
typedef __attribute__((ext_vector_type(8)))  float  v8f;

__device__ __forceinline__ unsigned short f2bf(float f) {
  unsigned int u = __float_as_uint(f);
  unsigned int r = u + 0x7FFFu + ((u >> 16) & 1u);   // round-to-nearest-even
  return (unsigned short)(r >> 16);
}

// ---------------------------------------------------------------------------
__global__ void k_zero(float* __restrict__ p, size_t n) {
  for (size_t i = (size_t)blockIdx.x * blockDim.x + threadIdx.x; i < n;
       i += (size_t)gridDim.x * blockDim.x)
    p[i] = 0.0f;
}

// ---------------------------------------------------------------------------
__global__ void k_cvt_bf16(const float* __restrict__ src,
                           unsigned short* __restrict__ dst, size_t n) {
  for (size_t i = (size_t)blockIdx.x * blockDim.x + threadIdx.x; i < n;
       i += (size_t)gridDim.x * blockDim.x)
    dst[i] = f2bf(src[i]);
}

// ---------------------------------------------------------------------------
// Pack W [din x dout] fp32 row-major into per-wave WMMA B fragments (bf16).
// Fragment (kt, nt): 32 lanes x 16 bf16; lane L, element e maps to
//   K = kt*32 + 8*(L>>4) + (e&7) + ((e>>3)<<4),  N = nt*16 + (L&15)
__global__ void k_pack_w(const float* __restrict__ W,
                         unsigned short* __restrict__ Wpk,
                         int din, int dout) {
  const int ntiles = dout >> 4;
  const size_t total = (size_t)din * dout;
  for (size_t idx = (size_t)blockIdx.x * blockDim.x + threadIdx.x; idx < total;
       idx += (size_t)gridDim.x * blockDim.x) {
    int e    = (int)(idx & 15);
    int lane = (int)((idx >> 4) & 31);
    size_t t = idx >> 9;
    int nt   = (int)(t % ntiles);
    int kt   = (int)(t / ntiles);
    int K = kt * 32 + 8 * (lane >> 4) + (e & 7) + ((e >> 3) << 4);
    int N = nt * 16 + (lane & 15);
    Wpk[idx] = f2bf(W[(size_t)K * dout + N]);
  }
}

// ---------------------------------------------------------------------------
// GEMM: H[N x dout] = Xbf[N x K] @ Wpk + bias.   K = 512.
// Block = 256 threads = 8 waves. Block tile: 80 rows x 128 cols.
// A tile (80 x 512 bf16 = 80KB) async-copied to LDS once; each wave computes
// 5 stacked 16x16 tiles for one 16-col N tile -> 5 WMMAs per B fragment.
__global__ __launch_bounds__(256)
void k_gemm_bf16(const unsigned short* __restrict__ Xbf,
                 const unsigned short* __restrict__ Wpk,
                 const float* __restrict__ bias,
                 float* __restrict__ H,
                 int K, int dout) {
  extern __shared__ unsigned short Atile[];        // MROWS x K bf16
  const int lane = threadIdx.x & 31;
  const int wave = threadIdx.x >> 5;
  const int m0   = blockIdx.x * MROWS;
  const int nt   = blockIdx.y * 8 + wave;
  const int n0   = nt * 16;
  const int half = lane >> 4;
  const int r    = lane & 15;
  const int ntiles = dout >> 4;
  const int ksteps = K >> 5;

  // ---- async copy of the contiguous A tile (MROWS*K*2 bytes) into LDS ----
  {
    const unsigned long long gbase =
        (unsigned long long)(const void*)(Xbf + (size_t)m0 * K);
    const unsigned lbase = (unsigned)(unsigned long long)(const void*)Atile;
    const int nchunks = (MROWS * 512 * 2) / 16 / 256;   // 20 x b128 per thread
#pragma unroll
    for (int i = 0; i < nchunks; ++i) {
      const unsigned boff = ((unsigned)threadIdx.x + (unsigned)i * 256u) * 16u;
      const unsigned      ldsa = lbase + boff;
      const unsigned long long ga = gbase + boff;
      asm volatile("global_load_async_to_lds_b128 %0, %1, off"
                   :: "v"(ldsa), "v"(ga) : "memory");
    }
    asm volatile("s_wait_asynccnt 0x0" ::: "memory");
  }
  __syncthreads();

  const unsigned short* bp = Wpk + ((size_t)nt * 32 + lane) * 16;
  const size_t bstride = (size_t)ntiles * 512;

  v8f acc[MTILES];
#pragma unroll
  for (int mt = 0; mt < MTILES; ++mt) acc[mt] = (v8f)0.0f;

  for (int kt = 0; kt < ksteps; ++kt) {
    v16bf b = *reinterpret_cast<const v16bf*>(bp);
    __builtin_prefetch(bp + bstride, 0, 1);        // global_prefetch_b8
    bp += bstride;

    // Preload all A fragments first (clause of ds_load_b128, one wait) ...
    v16bf a[MTILES];
#pragma unroll
    for (int mt = 0; mt < MTILES; ++mt) {
      const unsigned short* ap =
          Atile + (size_t)(mt * 16 + r) * K + 8 * half + kt * 32;
      v8bf alo = *reinterpret_cast<const v8bf*>(ap);        // ds_load_b128
      v8bf ahi = *reinterpret_cast<const v8bf*>(ap + 16);   // ds_load_b128
      a[mt] = __builtin_shufflevector(alo, ahi, 0, 1, 2, 3, 4, 5, 6, 7,
                                      8, 9, 10, 11, 12, 13, 14, 15);
    }
    // ... then 5 hazard-free back-to-back WMMAs (disjoint accumulators).
#pragma unroll
    for (int mt = 0; mt < MTILES; ++mt) {
      acc[mt] = __builtin_amdgcn_wmma_f32_16x16x32_bf16(
          false, a[mt], false, b, (short)0, acc[mt], false, false);
    }
  }

  // C/D layout: VGPR j -> row base + j + 8*half, col n0 + r
  const int col = n0 + r;
  const float bb = bias[col];
#pragma unroll
  for (int mt = 0; mt < MTILES; ++mt) {
    float* outp = H + (size_t)(m0 + mt * 16 + 8 * half) * dout + col;
#pragma unroll
    for (int j = 0; j < 8; ++j)
      outp[(size_t)j * dout] = acc[mt][j] + bb;
  }
}

// ---------------------------------------------------------------------------
// SPMM scatter: AG[dst] += w * H[src].  D/4 threads per edge, float4 each.
__global__ __launch_bounds__(256)
void k_spmm(const int* __restrict__ src, const int* __restrict__ dst,
            const float* __restrict__ ew, const float* __restrict__ H,
            float* __restrict__ AG, int E, int D) {
  const int tpe = D >> 2;
  const int epb = blockDim.x / tpe;
  const int e = blockIdx.x * epb + (int)(threadIdx.x / tpe);
  if (e >= E) return;
  const int c4 = ((int)(threadIdx.x % tpe)) << 2;
  const int s = src[e], d = dst[e];
  const float w = ew[e];
  const float4 hv = *reinterpret_cast<const float4*>(H + (size_t)s * D + c4);
  float* o = AG + (size_t)d * D + c4;
  atomicAdd(o + 0, w * hv.x);
  atomicAdd(o + 1, w * hv.y);
  atomicAdd(o + 2, w * hv.z);
  atomicAdd(o + 3, w * hv.w);
}

// ---------------------------------------------------------------------------
__global__ __launch_bounds__(512)
void k_colsum(const float* __restrict__ AG, float* __restrict__ colsum,
              int N, int D) {
  const int c = threadIdx.x;
  float s = 0.0f;
  for (int rr = blockIdx.x; rr < N; rr += gridDim.x)
    s += AG[(size_t)rr * D + c];
  atomicAdd(&colsum[c], s);
}

__global__ __launch_bounds__(512)
void k_sumsq(const float* __restrict__ AG, const float* __restrict__ colsum,
             float* __restrict__ out, int N, int D) {
  __shared__ float red[512];
  const int c = threadIdx.x;
  const float mean = colsum[c] * (1.0f / (float)N);
  float s = 0.0f;
  for (int rr = blockIdx.x; rr < N; rr += gridDim.x) {
    float v = AG[(size_t)rr * D + c] - mean;
    s += v * v;
  }
  red[c] = s;
  __syncthreads();
  for (int off = 256; off > 0; off >>= 1) {
    if (c < off) red[c] += red[c + off];
    __syncthreads();
  }
  if (c == 0) atomicAdd(out, red[0]);
}

// ---------------------------------------------------------------------------
__global__ void k_pn_relu_res(const float* __restrict__ AG,
                              const float* __restrict__ colsum,
                              const float* __restrict__ sumsq,
                              const float* __restrict__ xold,
                              float* __restrict__ xnew,
                              unsigned short* __restrict__ xbf,
                              int N, int D, int has_res) {
  const float invN = 1.0f / (float)N;
  const float rs = rsqrtf(1e-6f + sumsq[0] * invN);
  const size_t total = (size_t)N * D;
  for (size_t i = (size_t)blockIdx.x * blockDim.x + threadIdx.x; i < total;
       i += (size_t)gridDim.x * blockDim.x) {
    const int c = (int)(i & (size_t)(D - 1));
    float v = (AG[i] - colsum[c] * invN) * rs;
    v = fmaxf(v, 0.0f);
    if (has_res) v += xold[i];
    xnew[i] = v;
    xbf[i] = f2bf(v);
  }
}

// ---------------------------------------------------------------------------
extern "C" void kernel_launch(void* const* d_in, const int* in_sizes, int n_in,
                              void* d_out, int out_size, void* d_ws, size_t ws_size,
                              hipStream_t stream) {
  (void)in_sizes; (void)n_in; (void)out_size; (void)ws_size;

  const float* x    = (const float*)d_in[0];
  const float* ew   = (const float*)d_in[1];
  const float* Wl[3]  = {(const float*)d_in[2], (const float*)d_in[4], (const float*)d_in[6]};
  const float* bl[3]  = {(const float*)d_in[3], (const float*)d_in[5], (const float*)d_in[7]};
  const float* Wout = (const float*)d_in[8];
  const float* bout = (const float*)d_in[9];
  const int* esrc = (const int*)d_in[10];
  const int* edst = (const int*)d_in[11];
  float* out = (float*)d_out;

  // ---- workspace carve-up (all offsets 256B aligned) ----
  char* ws = (char*)d_ws;
  size_t off = 0;
  unsigned short* Xbf = (unsigned short*)(ws + off); off += (size_t)NNODES * DFEAT * 2;
  float* Xres = (float*)(ws + off);                  off += (size_t)NNODES * DFEAT * 4;
  float* H    = (float*)(ws + off);                  off += (size_t)NNODES * DFEAT * 4;
  float* AG   = (float*)(ws + off);                  off += (size_t)NNODES * DFEAT * 4;
  unsigned short* Wpk[3];
  for (int l = 0; l < 3; ++l) { Wpk[l] = (unsigned short*)(ws + off); off += (size_t)DFEAT * DFEAT * 2; }
  unsigned short* Wpko = (unsigned short*)(ws + off); off += (size_t)DFEAT * DCLS * 2;
  float* colsum = (float*)(ws + off);                off += 512 * 4;
  float* sumsq  = (float*)(ws + off);                off += 256;

  // ---- weight packing + input conversion ----
  for (int l = 0; l < 3; ++l)
    k_pack_w<<<512, 256, 0, stream>>>(Wl[l], Wpk[l], DFEAT, DFEAT);
  k_pack_w<<<512, 256, 0, stream>>>(Wout, Wpko, DFEAT, DCLS);
  k_cvt_bf16<<<8192, 256, 0, stream>>>(x, Xbf, (size_t)NNODES * DFEAT);

  const size_t lds_bytes = (size_t)MROWS * DFEAT * 2;     // 80 KB A tile
  const dim3 gemm_grid(NNODES / MROWS, DFEAT / 128);      // 625 x 4

  for (int l = 0; l < 3; ++l) {
    k_gemm_bf16<<<gemm_grid, 256, lds_bytes, stream>>>(Xbf, Wpk[l], bl[l], H,
                                                       DFEAT, DFEAT);
    k_zero<<<8192, 256, 0, stream>>>(AG, (size_t)NNODES * DFEAT);
    k_zero<<<1, 256, 0, stream>>>(colsum, 513);
    k_spmm<<<NEDGES / 2, 256, 0, stream>>>(esrc, edst, ew, H, AG, NEDGES, DFEAT);
    k_colsum<<<256, 512, 0, stream>>>(AG, colsum, NNODES, DFEAT);
    k_sumsq<<<256, 512, 0, stream>>>(AG, colsum, sumsq, NNODES, DFEAT);
    k_pn_relu_res<<<8192, 256, 0, stream>>>(AG, colsum, sumsq, Xres, Xres, Xbf,
                                            NNODES, DFEAT, l > 0 ? 1 : 0);
  }

  const dim3 gemm_grid_o(NNODES / MROWS, DCLS / 128);     // 625 x 2
  k_gemm_bf16<<<gemm_grid_o, 256, lds_bytes, stream>>>(Xbf, Wpko, bout, H,
                                                       DFEAT, DCLS);
  k_zero<<<8192, 256, 0, stream>>>(out, (size_t)NNODES * DCLS);
  k_spmm<<<NEDGES / 4, 256, 0, stream>>>(esrc, edst, ew, H, out, NEDGES, DCLS);
}
